// DiscreteStateTransition_86294482912149
// MI455X (gfx1250) — matrix operand order repaired and checked by reading, not compiled, not run on hardware
//
#include <hip/hip_runtime.h>
#include <hip/hip_bf16.h>

// ---------------------------------------------------------------------------
// Fused GNN message-passing block for MI455X (gfx1250, wave32, WMMA bf16).
// v2: single-pass M=64 (no edge-group loop) -> each register-resident W1T/W2T
// B-fragment is reused across 4 M-tiles; 3 fewer barriers; half the global
// weight traffic per block. LDS ~97KB (gfx1250 allows up to 320KB/WG).
// ---------------------------------------------------------------------------

typedef __bf16 bf16;
typedef __attribute__((ext_vector_type(16))) __bf16 v16bf;
typedef __attribute__((ext_vector_type(8)))  __bf16 v8bf;
typedef __attribute__((ext_vector_type(8)))  float  v8f;

#define B_DIM 4
#define O_DIM 8
#define T_DIM 512

// A-fragment (16x32 bf16): per lane two 8-half chunks at K = base + (lane/16)*8
// and K = base + 16 + (lane/16)*8 (caller pre-adds (lane/16)*8 into p).
__device__ __forceinline__ v16bf ld_a_frag(const bf16* p) {
  v8bf lo = *(const v8bf*)(p);
  v8bf hi = *(const v8bf*)(p + 16);
  v16bf r;
#pragma unroll
  for (int i = 0; i < 8; ++i) { r[i] = lo[i]; r[i + 8] = hi[i]; }
  return r;
}

__device__ __forceinline__ v8f wmma_bf16(v16bf a, v16bf b, v8f c) {
  // D = A(16x32) * B(32x16) + C, f32 accumulate
  return __builtin_amdgcn_wmma_f32_16x16x32_bf16(false, a, false, b,
                                                 (short)0, c, false, false);
}

__device__ __forceinline__ v8f vzero8() {
  v8f a;
#pragma unroll
  for (int i = 0; i < 8; ++i) a[i] = 0.0f;
  return a;
}

// ---------------------------------------------------------------------------
// Pack kernel: f32 weights -> bf16, transposed to [N][K] so B-fragments are
// two contiguous 16B chunks per lane.
//   W1 (128x512) -> W1T [512][128]
//   W2 (512x512) -> W2T [512][512]
//   Wn (576x 64) -> WnT [ 64][576]
// ---------------------------------------------------------------------------
__global__ __launch_bounds__(256) void pack_weights_kernel(
    const float* __restrict__ W1, const float* __restrict__ W2,
    const float* __restrict__ Wn,
    bf16* __restrict__ W1T, bf16* __restrict__ W2T, bf16* __restrict__ WnT) {
  int idx = blockIdx.x * 256 + threadIdx.x;
  if (idx < 512 * 128) {
    int n = idx >> 7, k = idx & 127;
    W1T[idx] = (bf16)W1[k * 512 + n];
  } else if (idx < 512 * 128 + 512 * 512) {
    int i = idx - 512 * 128;
    int n = i >> 9, k = i & 511;
    W2T[i] = (bf16)W2[k * 512 + n];
  } else if (idx < 512 * 128 + 512 * 512 + 64 * 576) {
    int i = idx - (512 * 128 + 512 * 512);
    int n = i / 576, k = i - n * 576;
    WnT[i] = (bf16)Wn[k * 64 + n];
  }
}

// ---------------------------------------------------------------------------
// Main fused kernel: one 256-thread block (8 waves) per (b,t).
// LDS map (97 KB):
//   [     0,  1024)  s_in      bf16 [8][64]
//   [  1024, 17408)  s_agg     f32  [8][512]
//   [ 17408, 33792)  s_premsg  bf16 [64][128]  \ overlaid later by
//   [ 33792, 99328)  s_msg1    bf16 [64][512]  /  s_netin bf16 [16][576]
// ---------------------------------------------------------------------------
__global__ __launch_bounds__(256) void fused_gnn_kernel(
    const float* __restrict__ x,   // (B,O,T,56)
    const float* __restrict__ fp,  // (B,O,T,8)
    const float* __restrict__ b1,  // (512)
    const float* __restrict__ b2,  // (512)
    const float* __restrict__ bn,  // (64)
    const bf16* __restrict__ W1T,  // [512][128]
    const bf16* __restrict__ W2T,  // [512][512]
    const bf16* __restrict__ WnT,  // [64][576]
    float* __restrict__ out)       // (B,O,T,64)
{
  __shared__ __align__(16) char smem[99328];
  bf16*  s_in     = (bf16*)(smem);
  float* s_agg    = (float*)(smem + 1024);
  bf16*  s_premsg = (bf16*)(smem + 17408);
  bf16*  s_msg1   = (bf16*)(smem + 33792);
  bf16*  s_netin  = (bf16*)(smem + 17408);  // overlay, used after GEMM2

  const int tid  = threadIdx.x;
  const int lane = tid & 31;
  const int wave = tid >> 5;
  const int nh   = lane & 15;  // A row within tile / B,C column within tile
  const int kh   = lane >> 4;  // K-half select

  const int blk = blockIdx.x;
  const int b = blk / T_DIM;
  const int t = blk - b * T_DIM;

  // ---- stage 0: load node features (concat x, probs) as bf16; zero agg ----
  for (int idx = tid; idx < 8 * 64; idx += 256) {
    int o = idx >> 6, c = idx & 63;
    size_t base = (size_t)(b * 8 + o) * T_DIM + t;
    float v = (c < 56) ? x[base * 56 + c] : fp[base * 8 + (c - 56)];
    s_in[idx] = (bf16)v;
  }
  for (int idx = tid; idx < 8 * 512; idx += 256) s_agg[idx] = 0.0f;

  // build pre_msg rows: [recv(e) features | send(e) features], 56 edges + pad
  __syncthreads();
  for (int idx = tid; idx < 64 * 8; idx += 256) {  // 64 rows x 8 segs of 16
    int r = idx >> 3, seg = idx & 7;
    bf16* dst = s_premsg + r * 128 + seg * 16;
    if (r < 56) {
      int i = r / 7, jp = r - i * 7;
      int recvn = jp + (jp >= i ? 1 : 0);
      int node = (seg < 4) ? recvn : i;
      const bf16* src = s_in + node * 64 + (seg & 3) * 16;
#pragma unroll
      for (int q = 0; q < 16; ++q) dst[q] = src[q];
    } else {
#pragma unroll
      for (int q = 0; q < 16; ++q) dst[q] = (bf16)0.0f;
    }
  }
  __syncthreads();

  // ---- GEMM1: (64x128) @ W1T -> relu -> s_msg1 (64x512 bf16) ----
  // 4 M-tiles x 32 N-tiles; wave w owns N-tiles [4w, 4w+4).
  // B-fragments (16 per wave) stay register-resident across all 4 M-tiles.
  for (int mt = 0; mt < 4; ++mt) {
    v8f acc[4];
#pragma unroll
    for (int j = 0; j < 4; ++j) acc[j] = vzero8();
    const bf16* arow = s_premsg + (mt * 16 + nh) * 128 + kh * 8;
#pragma unroll
    for (int kk = 0; kk < 4; ++kk) {
      v16bf a = ld_a_frag(arow + kk * 32);
#pragma unroll
      for (int j = 0; j < 4; ++j) {
        int n = (wave * 4 + j) * 16 + nh;
        v16bf bb = ld_a_frag(W1T + n * 128 + kk * 32 + kh * 8);
        acc[j] = wmma_bf16(a, bb, acc[j]);
      }
    }
    int mbase = mt * 16 + kh * 8;
#pragma unroll
    for (int j = 0; j < 4; ++j) {
      int n = (wave * 4 + j) * 16 + nh;
      float bias = b1[n];
#pragma unroll
      for (int q = 0; q < 8; ++q) {
        float v = acc[j][q] + bias;
        v = v > 0.0f ? v : 0.0f;
        s_msg1[(mbase + q) * 512 + n] = (bf16)v;
      }
    }
  }
  __syncthreads();

  // ---- GEMM2: (64x512) @ W2T -> relu -> scatter-add into s_agg ----
  // 64 B-fragments per wave preloaded by compiler; reused across 4 M-tiles.
  for (int mt = 0; mt < 4; ++mt) {
    v8f acc[4];
#pragma unroll
    for (int j = 0; j < 4; ++j) acc[j] = vzero8();
    const bf16* arow = s_msg1 + (mt * 16 + nh) * 512 + kh * 8;
#pragma unroll
    for (int kk = 0; kk < 16; ++kk) {
      v16bf a = ld_a_frag(arow + kk * 32);
#pragma unroll
      for (int j = 0; j < 4; ++j) {
        int n = (wave * 4 + j) * 16 + nh;
        v16bf bb = ld_a_frag(W2T + n * 512 + kk * 32 + kh * 8);
        acc[j] = wmma_bf16(a, bb, acc[j]);
      }
    }
    int mbase = mt * 16 + kh * 8;
#pragma unroll
    for (int j = 0; j < 4; ++j) {
      int n = (wave * 4 + j) * 16 + nh;
      float bias = b2[n];
#pragma unroll
      for (int q = 0; q < 8; ++q) {
        int e = mbase + q;
        if (e < 56) {
          float v = acc[j][q] + bias;
          v = v > 0.0f ? v : 0.0f;
          int i = e / 7, jp = e - i * 7;
          int o = jp + (jp >= i ? 1 : 0);           // recv node
          atomicAdd(&s_agg[o * 512 + n], v);        // ds_add_f32
        }
      }
    }
  }
  __syncthreads();

  // ---- build net_in (16x576 bf16, rows 8..15 zero pad) over dead buffers ----
  for (int idx = tid; idx < 16 * 576; idx += 256) {
    int o = idx / 576;
    int c = idx - o * 576;
    bf16 v;
    if (o < 8)
      v = (c < 64) ? s_in[o * 64 + c] : (bf16)s_agg[o * 512 + (c - 64)];
    else
      v = (bf16)0.0f;
    s_netin[idx] = v;
  }
  __syncthreads();

  // ---- GEMM3: (16x576) @ WnT (64x576) -> out; 4 waves, one N-tile each ----
  if (wave < 4) {
    v8f acc = vzero8();
    const bf16* arow = s_netin + nh * 576 + kh * 8;
    const bf16* brow = WnT + (wave * 16 + nh) * 576 + kh * 8;
#pragma unroll
    for (int kk = 0; kk < 18; ++kk) {
      v16bf a = ld_a_frag(arow + kk * 32);
      v16bf bb = ld_a_frag(brow + kk * 32);
      acc = wmma_bf16(a, bb, acc);
    }
    if (lane < 16) {                 // lanes 0-15 hold M=0..7 (real nodes)
      int n = wave * 16 + lane;
      float bias = bn[n];
#pragma unroll
      for (int q = 0; q < 8; ++q) {
        out[(((size_t)b * 8 + q) * T_DIM + t) * 64 + n] = acc[q] + bias;
      }
    }
  }
}

// ---------------------------------------------------------------------------
// Launch: pack weights into d_ws (bf16 transposed), then fused kernel.
// ws layout: W1T @0 (128KB) | W2T @131072 (512KB) | WnT @655360 (72KB)
// ---------------------------------------------------------------------------
extern "C" void kernel_launch(void* const* d_in, const int* in_sizes, int n_in,
                              void* d_out, int out_size, void* d_ws, size_t ws_size,
                              hipStream_t stream) {
  (void)in_sizes; (void)n_in; (void)out_size; (void)ws_size;
  const float* x  = (const float*)d_in[1];
  const float* fp = (const float*)d_in[3];
  const float* W1 = (const float*)d_in[6];
  const float* b1 = (const float*)d_in[7];
  const float* W2 = (const float*)d_in[8];
  const float* b2 = (const float*)d_in[9];
  const float* Wn = (const float*)d_in[10];
  const float* bn = (const float*)d_in[11];

  bf16* W1T = (bf16*)d_ws;
  bf16* W2T = (bf16*)((char*)d_ws + 131072);
  bf16* WnT = (bf16*)((char*)d_ws + 655360);

  const int pack_elems = 512 * 128 + 512 * 512 + 64 * 576;
  pack_weights_kernel<<<(pack_elems + 255) / 256, 256, 0, stream>>>(
      W1, W2, Wn, W1T, W2T, WnT);

  fused_gnn_kernel<<<B_DIM * T_DIM, 256, 0, stream>>>(
      x, fp, b1, b2, bn, W1T, W2T, WnT, (float*)d_out);
}
